// MultiHeadCrossAttention_51075751084251
// MI455X (gfx1250) — matrix operand order repaired and checked by reading, not compiled, not run on hardware
//
#include <hip/hip_runtime.h>
#include <hip/hip_bf16.h>
#include <math.h>

#define B_  2
#define T_  2048
#define E_  1024
#define H_  16
#define D_  64
#define SCALE 0.125f   // 1/sqrt(64)

typedef __attribute__((ext_vector_type(16))) __bf16 v16bf;
typedef __attribute__((ext_vector_type(8)))  __bf16 v8bf;
typedef __attribute__((ext_vector_type(8)))  float  v8f;

static __device__ __forceinline__ __bf16 to_bf16(float f) {
  unsigned u = __float_as_uint(f);
  u += 0x7FFFu + ((u >> 16) & 1u);          // round-to-nearest-even
  unsigned short h = (unsigned short)(u >> 16);
  __bf16 r; __builtin_memcpy(&r, &h, 2);
  return r;
}

static __device__ __forceinline__ v8f zero8() {
  v8f z = {0.f,0.f,0.f,0.f,0.f,0.f,0.f,0.f};
  return z;
}

static __device__ __forceinline__ v8f wmma_bf16(v16bf a, v16bf b, v8f c) {
  return __builtin_amdgcn_wmma_f32_16x16x32_bf16(false, a, false, b, (short)0, c, false, false);
}

// A fragment (16x32 bf16, lane holds row m=lane&15):
// K elements = {hf*8 .. hf*8+7} U {16+hf*8 .. 16+hf*8+7}  -> two b128 loads
static __device__ __forceinline__ v16bf load_a_frag(const __bf16* row) {
  const int hf = (threadIdx.x & 31) >> 4;
  v8bf lo = *(const v8bf*)(row + hf * 8);
  v8bf hi = *(const v8bf*)(row + 16 + hf * 8);
  v16bf f;
#pragma unroll
  for (int i = 0; i < 8; ++i) { f[i] = lo[i]; f[i + 8] = hi[i]; }
  return f;
}

// B fragment (32x16 bf16, lane holds column n=lane&15 of B == row n of B^T):
// K elements = {hf*8 .. hf*8+7} U {16+hf*8 .. 16+hf*8+7}  -> two b128 loads
static __device__ __forceinline__ v16bf load_b_frag(const __bf16* btrow) {
  const int hf = (threadIdx.x & 31) >> 4;
  v8bf lo = *(const v8bf*)(btrow + hf * 8);
  v8bf hi = *(const v8bf*)(btrow + 16 + hf * 8);
  v16bf f;
#pragma unroll
  for (int i = 0; i < 8; ++i) { f[i] = lo[i]; f[i + 8] = hi[i]; }
  return f;
}

// ---------------- elementwise / layout prep kernels ----------------

__global__ void f32_to_bf16_k(const float* __restrict__ in, __bf16* __restrict__ out, int n) {
  int i = blockIdx.x * blockDim.x + threadIdx.x;
  if (i < n) out[i] = to_bf16(in[i]);
}

// W [K][N] f32 (row-major)  ->  WT [N][K] bf16 (K contiguous per output row)
__global__ void transpose_w_k(const float* __restrict__ W, __bf16* __restrict__ WT, int K, int N) {
  int i = blockIdx.x * blockDim.x + threadIdx.x;
  if (i >= K * N) return;
  int k = i / N, n = i % N;
  WT[(size_t)n * K + k] = to_bf16(W[i]);
}

// V part of KV buffer -> Vt[(bh*D + d)*T + k]  (k contiguous)
__global__ void transpose_v_k(const __bf16* __restrict__ KV, __bf16* __restrict__ Vt) {
  int i = blockIdx.x * blockDim.x + threadIdx.x;
  if (i >= B_ * H_ * D_ * T_) return;
  int k   = i % T_;
  int rem = i / T_;
  int d   = rem % D_;
  int bh  = rem / D_;
  int b = bh / H_, h = bh % H_;
  Vt[i] = KV[(size_t)(b * T_ + k) * (2 * E_) + E_ + h * D_ + d];
}

// ---------------- GEMM: out = A(bf16) @ B + bias, B given as B^T (bf16) ----------------
// block = 256 thr = 8 waves; block tile 256(M) x 64(N); each wave: 32 rows x 4 n-strips
// (each B fragment feeds 2 WMMAs)

template <bool OUT_BF16>
__global__ void __launch_bounds__(256)
gemm_bias(const __bf16* __restrict__ A, const __bf16* __restrict__ BT,
          const float* __restrict__ bias, void* __restrict__ out,
          int M, int N, int K) {
  const int lane = threadIdx.x & 31;
  const int wave = threadIdx.x >> 5;
  const int lm   = lane & 15;
  const int hf   = lane >> 4;

  const int row0 = blockIdx.y * 256 + wave * 32;
  const int n0   = blockIdx.x * 64;

  const __bf16* arow0 = A + (size_t)(row0 + lm) * K;
  const __bf16* arow1 = A + (size_t)(row0 + 16 + lm) * K;
  const __bf16* bt[4];
#pragma unroll
  for (int s = 0; s < 4; ++s) bt[s] = BT + (size_t)(n0 + s * 16 + lm) * K;

  v8f acc[2][4];
#pragma unroll
  for (int i = 0; i < 2; ++i)
#pragma unroll
    for (int s = 0; s < 4; ++s) acc[i][s] = zero8();

  for (int kb = 0; kb < K; kb += 32) {
    v16bf a0 = load_a_frag(arow0 + kb);
    v16bf a1 = load_a_frag(arow1 + kb);
#pragma unroll
    for (int s = 0; s < 4; ++s) {
      v16bf b = load_b_frag(bt[s] + kb);
      acc[0][s] = wmma_bf16(a0, b, acc[0][s]);
      acc[1][s] = wmma_bf16(a1, b, acc[1][s]);
    }
  }

#pragma unroll
  for (int s = 0; s < 4; ++s) {
    int n = n0 + s * 16 + lm;
    float bv = bias[n];
#pragma unroll
    for (int i = 0; i < 2; ++i) {
#pragma unroll
      for (int r = 0; r < 8; ++r) {
        float v = acc[i][s][r] + bv;
        size_t idx = (size_t)(row0 + i * 16 + 8 * hf + r) * N + n;
        if (OUT_BF16) ((__bf16*)out)[idx] = to_bf16(v);
        else          ((float*)out)[idx]  = v;
      }
    }
  }
}

// ---------------- Pass 1: per-column (query-axis) softmax stats ----------------
// grid (T/256, B*H), block 128 = 4 waves; each wave owns a 64-wide k strip
// (8 K-fragments pinned in registers), streams all q tiles once,
// maintains online (max, sum-exp) per column.

__global__ void __launch_bounds__(128)
col_softmax_stats(const __bf16* __restrict__ Qb, const __bf16* __restrict__ KVb,
                  float* __restrict__ Ms, float* __restrict__ Ls) {
  const int lane = threadIdx.x & 31;
  const int wave = threadIdx.x >> 5;
  const int lm   = lane & 15;
  const int bh = blockIdx.y, b = bh / H_, h = bh % H_;
  const int kbase = blockIdx.x * 256 + wave * 64;

  v16bf bk[4][2];
#pragma unroll
  for (int s = 0; s < 4; ++s) {
    const __bf16* krow = KVb + (size_t)(b * T_ + kbase + s * 16 + lm) * (2 * E_) + h * D_;
    bk[s][0] = load_b_frag(krow);        // d = 0..31
    bk[s][1] = load_b_frag(krow + 32);   // d = 32..63
  }

  float m_run[4], l_run[4];
#pragma unroll
  for (int s = 0; s < 4; ++s) { m_run[s] = -3.0e38f; l_run[s] = 0.f; }

  for (int q0 = 0; q0 < T_; q0 += 16) {
    const __bf16* qrow = Qb + (size_t)(b * T_ + q0 + lm) * E_ + h * D_;
    v16bf a0 = load_a_frag(qrow);
    v16bf a1 = load_a_frag(qrow + 32);
#pragma unroll
    for (int s = 0; s < 4; ++s) {
      v8f c = zero8();
      c = wmma_bf16(a0, bk[s][0], c);
      c = wmma_bf16(a1, bk[s][1], c);

      float tm = -3.0e38f;
#pragma unroll
      for (int r = 0; r < 8; ++r) tm = fmaxf(tm, c[r] * SCALE);
      tm = fmaxf(tm, __shfl_xor(tm, 16, 32));   // combine the two lane-halves (same column)
      float mn = fmaxf(m_run[s], tm);
      float p = 0.f;
#pragma unroll
      for (int r = 0; r < 8; ++r) p += __expf(c[r] * SCALE - mn);
      p += __shfl_xor(p, 16, 32);
      l_run[s] = l_run[s] * __expf(m_run[s] - mn) + p;
      m_run[s] = mn;
    }
  }
  if ((lane >> 4) == 0) {
#pragma unroll
    for (int s = 0; s < 4; ++s) {
      Ms[(size_t)bh * T_ + kbase + s * 16 + lm] = m_run[s];
      Ls[(size_t)bh * T_ + kbase + s * 16 + lm] = l_run[s];
    }
  }
}

// ---------------- Pass 2: O = softmax_col(S) @ V, flash-style recompute ----------------
// grid (T/128, B*H), block 128 = 4 waves; each wave owns a 32-row q tile,
// streams k in steps of 32: S tiles (WMMA) -> P = exp(S-m[k])/l[k] -> LDS transpose
// (C-layout -> A-layout) -> O += P @ V (WMMA, Vt pre-transposed k-contiguous).
// Each K/V fragment now feeds 2 WMMAs (two q sub-tiles).

__global__ void __launch_bounds__(128)
attn_out(const __bf16* __restrict__ Qb, const __bf16* __restrict__ KVb,
         const __bf16* __restrict__ Vt, const float* __restrict__ Ms,
         const float* __restrict__ Ls, __bf16* __restrict__ Ob) {
  __shared__ __bf16 Plds[4][32][32];

  const int lane = threadIdx.x & 31;
  const int wave = threadIdx.x >> 5;
  const int lm   = lane & 15;
  const int hf   = lane >> 4;
  const int bh = blockIdx.y, b = bh / H_, h = bh % H_;
  const int qbase = blockIdx.x * 128 + wave * 32;

  v16bf aq[2][2];
#pragma unroll
  for (int i = 0; i < 2; ++i) {
    const __bf16* qrow = Qb + (size_t)(b * T_ + qbase + i * 16 + lm) * E_ + h * D_;
    aq[i][0] = load_a_frag(qrow);
    aq[i][1] = load_a_frag(qrow + 32);
  }

  const float* msp = Ms + (size_t)bh * T_;
  const float* lsp = Ls + (size_t)bh * T_;
  const __bf16* vt[4];
#pragma unroll
  for (int s = 0; s < 4; ++s) vt[s] = Vt + (size_t)(bh * D_ + s * 16 + lm) * T_;

  v8f o[2][4];
#pragma unroll
  for (int i = 0; i < 2; ++i)
#pragma unroll
    for (int s = 0; s < 4; ++s) o[i][s] = zero8();

  for (int k0 = 0; k0 < T_; k0 += 32) {
    const __bf16* kr0 = KVb + (size_t)(b * T_ + k0 + lm) * (2 * E_) + h * D_;
    const __bf16* kr1 = kr0 + (size_t)16 * (2 * E_);
    v16bf bk0a = load_b_frag(kr0);
    v16bf bk0b = load_b_frag(kr0 + 32);
    v16bf bk1a = load_b_frag(kr1);
    v16bf bk1b = load_b_frag(kr1 + 32);

    float m0 = msp[k0 + lm],      l0 = lsp[k0 + lm];
    float m1 = msp[k0 + 16 + lm], l1 = lsp[k0 + 16 + lm];
    float rl0 = 1.f / l0, rl1 = 1.f / l1;

#pragma unroll
    for (int i = 0; i < 2; ++i) {
      v8f s0 = zero8(), s1 = zero8();
      s0 = wmma_bf16(aq[i][0], bk0a, s0);
      s0 = wmma_bf16(aq[i][1], bk0b, s0);
      s1 = wmma_bf16(aq[i][0], bk1a, s1);
      s1 = wmma_bf16(aq[i][1], bk1b, s1);
      // C layout: lane holds column lm, rows r + 8*hf -> scatter to LDS row-major P tile
#pragma unroll
      for (int r = 0; r < 8; ++r) {
        Plds[wave][i * 16 + r + 8 * hf][lm]      = to_bf16(__expf(s0[r] * SCALE - m0) * rl0);
        Plds[wave][i * 16 + r + 8 * hf][lm + 16] = to_bf16(__expf(s1[r] * SCALE - m1) * rl1);
      }
    }
    __syncthreads();
    v16bf ap0 = load_a_frag(&Plds[wave][lm][0]);        // A-layout P, q rows 0..15
    v16bf ap1 = load_a_frag(&Plds[wave][16 + lm][0]);   // A-layout P, q rows 16..31
#pragma unroll
    for (int s = 0; s < 4; ++s) {
      v16bf bv = load_b_frag(vt[s] + k0);
      o[0][s] = wmma_bf16(ap0, bv, o[0][s]);
      o[1][s] = wmma_bf16(ap1, bv, o[1][s]);
    }
    __syncthreads();
  }

#pragma unroll
  for (int i = 0; i < 2; ++i) {
#pragma unroll
    for (int s = 0; s < 4; ++s) {
#pragma unroll
      for (int r = 0; r < 8; ++r) {
        int q   = qbase + i * 16 + r + 8 * hf;
        int col = h * D_ + s * 16 + lm;
        Ob[(size_t)(b * T_ + q) * E_ + col] = to_bf16(o[i][s][r]);
      }
    }
  }
}

// ---------------- launch ----------------

extern "C" void kernel_launch(void* const* d_in, const int* in_sizes, int n_in,
                              void* d_out, int out_size, void* d_ws, size_t ws_size,
                              hipStream_t stream) {
  const float* input = (const float*)d_in[0];
  const float* enc   = (const float*)d_in[1];
  const float* Wq    = (const float*)d_in[2];
  const float* bq    = (const float*)d_in[3];
  const float* Wkv   = (const float*)d_in[4];
  const float* bkv   = (const float*)d_in[5];
  const float* Wo    = (const float*)d_in[6];
  const float* bo    = (const float*)d_in[7];

  char* ws = (char*)d_ws;
  size_t off = 0;
  auto alloc = [&](size_t bytes) -> char* {
    char* p = ws + off;
    off += (bytes + 255) & ~(size_t)255;
    return p;
  };

  const size_t MT = (size_t)B_ * T_;   // 4096 rows
  __bf16* Xbf  = (__bf16*)alloc(MT * E_ * 2);          // input  bf16
  __bf16* Ebf  = (__bf16*)alloc(MT * E_ * 2);          // encoded bf16
  __bf16* WqT  = (__bf16*)alloc((size_t)E_ * E_ * 2);
  __bf16* WkvT = (__bf16*)alloc((size_t)E_ * 2 * E_ * 2);
  __bf16* WoT  = (__bf16*)alloc((size_t)E_ * E_ * 2);
  __bf16* Qbf  = (__bf16*)alloc(MT * E_ * 2);
  __bf16* KVbf = (__bf16*)alloc(MT * 2 * E_ * 2);
  float*  Ms   = (float*)alloc((size_t)B_ * H_ * T_ * 4);
  float*  Ls   = (float*)alloc((size_t)B_ * H_ * T_ * 4);
  __bf16* Vt   = Ebf;   // alias: encoded bf16 dead after KV GEMM
  __bf16* Ob   = Xbf;   // alias: input bf16 dead after Q GEMM

  const int nconv = (int)(MT * E_);
  f32_to_bf16_k<<<(nconv + 255) / 256, 256, 0, stream>>>(input, Xbf, nconv);
  f32_to_bf16_k<<<(nconv + 255) / 256, 256, 0, stream>>>(enc,   Ebf, nconv);
  transpose_w_k<<<(E_ * E_ + 255) / 256, 256, 0, stream>>>(Wq,  WqT,  E_, E_);
  transpose_w_k<<<(E_ * 2 * E_ + 255) / 256, 256, 0, stream>>>(Wkv, WkvT, E_, 2 * E_);
  transpose_w_k<<<(E_ * E_ + 255) / 256, 256, 0, stream>>>(Wo,  WoT,  E_, E_);

  dim3 gq(E_ / 64, (int)(MT / 256));        // (16, 16)
  gemm_bias<true><<<gq, 256, 0, stream>>>(Xbf, WqT, bq, Qbf, (int)MT, E_, E_);
  dim3 gkv(2 * E_ / 64, (int)(MT / 256));   // (32, 16)
  gemm_bias<true><<<gkv, 256, 0, stream>>>(Ebf, WkvT, bkv, KVbf, (int)MT, 2 * E_, E_);

  const int nv = B_ * H_ * D_ * T_;
  transpose_v_k<<<(nv + 255) / 256, 256, 0, stream>>>(KVbf, Vt);

  dim3 gs(T_ / 256, B_ * H_);               // (8, 32)
  col_softmax_stats<<<gs, 128, 0, stream>>>(Qbf, KVbf, Ms, Ls);
  dim3 ga(T_ / 128, B_ * H_);               // (16, 32)
  attn_out<<<ga, 128, 0, stream>>>(Qbf, KVbf, Vt, Ms, Ls, Ob);

  gemm_bias<false><<<gq, 256, 0, stream>>>(Ob, WoT, bo, d_out, (int)MT, E_, E_);
}
